// StructurelLoss_31868657336725
// MI455X (gfx1250) — matrix operand order repaired and checked by reading, not compiled, not run on hardware
//
#include <hip/hip_runtime.h>
#include <hip/hip_bf16.h>
#include <math.h>

typedef __attribute__((ext_vector_type(16))) _Float16 v16h;
typedef __attribute__((ext_vector_type(8)))  float    v8f;

#define B       16           // batches
#define H       512
#define W       512
#define KER     31
#define PADR    15
#define BANDS   32           // 32 bands of 16 rows each
#define ROWS48  48           // 15 halo + 16 out + 15 halo, rounded to 3x16
#define MSTRIDE 576          // 16 left pad + 512 + 48 right pad (f16 elements)
#define LDS_M   (ROWS48 * MSTRIDE * 2)      // 55296 B  (f16 mask window)
#define LDS_H   (ROWS48 * W * 2)            // 49152 B  (f16 horizontal sums)
#define LDS_R   (256 * 4 * 4)               //  4096 B  (block reduction)
#define LDS_TOT (LDS_M + LDS_H + LDS_R)     // 108544 B dynamic LDS

// ---------------------------------------------------------------------------
// Main kernel: one block per (band, batch). Computes pooled via
// horizontal-box-sum-as-WMMA + vertical sliding sum, fuses the loss math, and
// writes 4 partial sums per block (deterministic tree reduction, no atomics).
// ---------------------------------------------------------------------------
__global__ __launch_bounds__(256)
void structure_loss_main(const float* __restrict__ pred,
                         const int*  __restrict__ mask,
                         float* __restrict__ partial /* [B][BANDS][4] */) {
    extern __shared__ char smem[];
    _Float16* bufM = (_Float16*)smem;                    // 48 x 576 mask window
    _Float16* bufH = (_Float16*)(smem + LDS_M);          // 48 x 512 horiz sums
    float*    red  = (float*)   (smem + LDS_M + LDS_H);  // 256 x 4 reduction

    const int band = blockIdx.x;          // 0..31
    const int b    = blockIdx.y;          // 0..15
    const int tid  = threadIdx.x;         // 0..255
    const int lane = tid & 31;
    const int wave = tid >> 5;

    // ---- Phase 1: stage zero-padded f16 mask window in LDS ----------------
    const int y0 = band * 16 - PADR;      // global row of local row 0
    for (int idx = tid; idx < ROWS48 * MSTRIDE; idx += 256) {
        const int r  = idx / MSTRIDE;
        const int cp = idx - r * MSTRIDE;
        const int x  = cp - 16;
        const int y  = y0 + r;
        float v = 0.0f;
        if ((unsigned)y < (unsigned)H && (unsigned)x < (unsigned)W) {
            int m = mask[((size_t)b * H + y) * W + x];
            if (m == 255) m = 0;
            v = (float)m;
        }
        bufM[idx] = (_Float16)v;
    }
    __syncthreads();

    // ---- Phase 2: horizontal 31-tap box sum as WMMA -----------------------
    // Output col n of a 16-wide tile sums inputs r in [n, n+30] relative to
    // input base (tile*16 - 15).  Split K into [0,32) and [32,64):
    //   B0[k][n] = (k >= n && k <= n+30),   B1[k][n] = (k <= n-2)
    const int n     = lane & 15;
    const int kbase = (lane >> 4) << 3;   // B layout: lanes 0-15 K=0-7,16-23; 16-31 K=8-15,24-31
    v16h B0mat, B1mat;
#pragma unroll
    for (int h = 0; h < 16; ++h) {
        const int k = kbase + h + ((h >= 8) ? 8 : 0);
        B0mat[h] = (_Float16)((k >= n && k <= n + 30) ? 1.0f : 0.0f);
        B1mat[h] = (_Float16)(((k + 32) <= n + 30) ? 1.0f : 0.0f);
    }

    const int mrow = lane & 15;           // A layout: lanes stripe M
    const int klo  = (lane >> 4) * 8;     // lanes 16-31 hold K offset +8

    for (int iter = 0; iter < 12; ++iter) {      // 96 jobs / 8 waves
        const int job = wave + (iter << 3);
        const int g   = job >> 5;                // row-group 0..2
        const int t   = job & 31;                // col tile  0..31
        const int row = (g << 4) + mrow;
        // padded col of K: 16 + (t*16 - 15) + K = t*16 + 1 + K
        const _Float16* rp = bufM + row * MSTRIDE + (t << 4) + 1;
        v16h A0, A1;
#pragma unroll
        for (int h = 0; h < 16; ++h) {
            const int K = h + ((h >= 8) ? 8 : 0) + klo;   // ISA 16-bit A 16x32
            A0[h] = rp[K];
            A1[h] = rp[K + 32];
        }
        v8f c = {};
        c = __builtin_amdgcn_wmma_f32_16x16x32_f16(false, A0, false, B0mat,
                                                   (short)0, c, false, false);
        c = __builtin_amdgcn_wmma_f32_16x16x32_f16(false, A1, false, B1mat,
                                                   (short)0, c, false, false);
        // D layout: lanes 0-15 -> M=v, N=lane; lanes 16-31 -> M=v+8, N=lane-16
        const int colb  = (t << 4) + n;
        const int rbase = (g << 4) + ((lane >> 4) << 3);
#pragma unroll
        for (int v = 0; v < 8; ++v)
            bufH[(rbase + v) * W + colb] = (_Float16)c[v];  // ints <= 961: exact
    }
    __syncthreads();

    // ---- Phase 3: vertical sliding sum + fused loss -----------------------
    float sw = 0.0f, swb = 0.0f, sint = 0.0f, scard = 0.0f;
#pragma unroll
    for (int cc = 0; cc < 2; ++cc) {
        const int x = tid + (cc << 8);
        float s = 0.0f;
#pragma unroll
        for (int r = 0; r < 31; ++r) s += (float)bufH[r * W + x];
        for (int i = 0; i < 16; ++i) {
            const int y = band * 16 + i;
            const float pooled = s * (1.0f / 961.0f);
            int m = mask[((size_t)b * H + y) * W + x];
            if (m == 255) m = 0;
            const float mf = (float)m;
            const float z0 = pred[(((size_t)b * 2 + 0) * H + y) * W + x];
            const float z1 = pred[(((size_t)b * 2 + 1) * H + y) * W + x];
            const float mx = fmaxf(z0, z1);
            const float lse = mx + log1pf(expf(fminf(z0, z1) - mx));
            const float lp0 = z0 - lse;
            const float lp1 = z1 - lse;
            const float wbce = -(m ? lp1 : lp0);
            const float p1   = expf(lp1);
            const float weit = 1.0f + 5.0f * fabsf(pooled - mf);
            sw    += weit;
            swb   += weit * wbce;
            sint  += p1 * mf * weit;
            scard += (p1 + mf) * weit;
            if (i < 15)
                s += (float)bufH[(i + 31) * W + x] - (float)bufH[i * W + x];
        }
    }

    // ---- Deterministic block reduction ------------------------------------
    red[tid * 4 + 0] = sw;
    red[tid * 4 + 1] = swb;
    red[tid * 4 + 2] = sint;
    red[tid * 4 + 3] = scard;
    for (int off = 128; off > 0; off >>= 1) {
        __syncthreads();
        if (tid < off) {
#pragma unroll
            for (int k = 0; k < 4; ++k)
                red[tid * 4 + k] += red[(tid + off) * 4 + k];
        }
    }
    if (tid == 0) {
#pragma unroll
        for (int k = 0; k < 4; ++k)
            partial[((size_t)b * BANDS + band) * 4 + k] = red[k];
    }
}

// ---------------------------------------------------------------------------
// Finalize: fixed-order sum of the 32 band partials per batch, compute
// per-batch wbce + wiou, then mean over 16 batches. Fully deterministic.
// ---------------------------------------------------------------------------
__global__ void structure_loss_finalize(const float* __restrict__ partial,
                                        float* __restrict__ out) {
    __shared__ float acc[16];
    const int t = threadIdx.x;
    if (t < 16) {
        float sw = 0.0f, swb = 0.0f, si = 0.0f, sc = 0.0f;
        for (int j = 0; j < BANDS; ++j) {
            const float* p = partial + ((size_t)t * BANDS + j) * 4;
            sw  += p[0];
            swb += p[1];
            si  += p[2];
            sc  += p[3];
        }
        const float wbce = swb / sw;
        const float uni  = sc - si;            // cardinality - inter
        const float wiou = 1.0f - (si + 1.0f) / (uni + 1.0f);
        acc[t] = wbce + wiou;
    }
    __syncthreads();
    if (t == 0) {
        float s = 0.0f;
        for (int j = 0; j < 16; ++j) s += acc[j];
        out[0] = s * (1.0f / 16.0f);
    }
}

extern "C" void kernel_launch(void* const* d_in, const int* in_sizes, int n_in,
                              void* d_out, int out_size, void* d_ws, size_t ws_size,
                              hipStream_t stream) {
    (void)in_sizes; (void)n_in; (void)out_size; (void)ws_size;
    const float* pred = (const float*)d_in[0];   // [16,2,512,512] f32
    const int*   mask = (const int*)d_in[1];     // [16,512,512] int
    float* partial = (float*)d_ws;               // [16][32][4] floats = 8 KB
    float* out = (float*)d_out;

    dim3 grid(BANDS, B);
    structure_loss_main<<<grid, 256, LDS_TOT, stream>>>(pred, mask, partial);
    structure_loss_finalize<<<1, 32, 0, stream>>>(partial, out);
}
